// ProgSGStyleModel_8821862826773
// MI455X (gfx1250) — matrix operand order repaired and checked by reading, not compiled, not run on hardware
//
#include <hip/hip_runtime.h>
#include <math.h>

#define HID 512
#define NHEADS 8
#define NLAYERS 4
#define NN 20000
#define NE 100000
#define NG 16
#define NTGT 4
#define INV_SQRT_C 0.125f

typedef __bf16 bf16_t;
typedef __attribute__((ext_vector_type(16))) __bf16 v16bf;
typedef __attribute__((ext_vector_type(8)))  __bf16 v8bf;
typedef __attribute__((ext_vector_type(8)))  float   v8f;

// ---------- device helpers ----------
__device__ inline bf16_t f2bf(float f) {
  unsigned u = __float_as_uint(f);
  unsigned r = (u + 0x7FFFu + ((u >> 16) & 1u)) >> 16;   // RNE
  unsigned short s = (unsigned short)r;
  return __builtin_bit_cast(bf16_t, s);
}
__device__ inline float eluf(float x) { return x > 0.f ? x : expm1f(x); }
__device__ inline unsigned encf(float f) {
  unsigned u = __float_as_uint(f);
  return (u & 0x80000000u) ? ~u : (u | 0x80000000u);
}
__device__ inline float decf(unsigned u) {
  return __uint_as_float((u & 0x80000000u) ? (u & 0x7FFFFFFFu) : ~u);
}
__device__ inline float wred(float v) {
#pragma unroll
  for (int o = 16; o; o >>= 1) v += __shfl_xor(v, o, 32);
  return v;
}

// ---------- utility kernels ----------
__global__ void k_fill_f32(float* p, float v, int n) {
  int t = blockIdx.x * 256 + threadIdx.x;
  if (t < n) p[t] = v;
}
__global__ void k_fill_u32(unsigned* p, unsigned v, int n) {
  int t = blockIdx.x * 256 + threadIdx.x;
  if (t < n) p[t] = v;
}
// W fp32 [K,Nout] row-major -> Wt bf16 [Nout,K]
__global__ void k_cvt_wT(const float* __restrict__ W, bf16_t* __restrict__ Wt,
                         int K, int Nout) {
  int t = blockIdx.x * 256 + threadIdx.x;
  if (t >= K * Nout) return;
  int n = t / K, k = t - n * K;
  Wt[t] = f2bf(W[(size_t)k * Nout + n]);
}
__global__ void k_cvt_bf16(const float* __restrict__ in, bf16_t* __restrict__ out, int n) {
  int t = blockIdx.x * 256 + threadIdx.x;
  if (t < n) out[t] = f2bf(in[t]);
}

// ---------- embeddings ----------
__global__ void k_embed_nodes(const int* __restrict__ xn, const float* __restrict__ emb,
                              float* __restrict__ h, bf16_t* __restrict__ hb) {
  int t = blockIdx.x * 256 + threadIdx.x;
  if (t >= NN * HID) return;
  int n = t >> 9, d = t & 511;
  float acc = 0.f;
#pragma unroll
  for (int f = 0; f < 3; ++f) {
    int idx = xn[n * 3 + f];
    acc += emb[((size_t)f * 120 + idx) * HID + d];
  }
  h[t] = acc;
  hb[t] = f2bf(acc);
}
__global__ void k_embed_edges(const int* __restrict__ ea, const float* __restrict__ emb,
                              bf16_t* __restrict__ eb) {
  long t = (long)blockIdx.x * 256 + threadIdx.x;
  if (t >= (long)NE * HID) return;
  int e = (int)(t >> 9), d = (int)(t & 511);
  float acc = emb[(size_t)(0 * 16 + ea[e * 2 + 0]) * HID + d] +
              emb[(size_t)(1 * 16 + ea[e * 2 + 1]) * HID + d];
  eb[t] = f2bf(acc);
}

// ---------- WMMA GEMM: C[M,Nout] = A[M,512] @ W[512,Nout] (+bias, +elu) ----------
// A: bf16 row-major [M,512]; Wt: bf16 transposed [Nout,512].
// One wave computes a 16x64 strip: A fragment reused across 4 WMMA accumulators.
__global__ __launch_bounds__(256) void k_gemm_bf16(
    const bf16_t* __restrict__ A, const bf16_t* __restrict__ Wt,
    const float* __restrict__ bias, float* __restrict__ C,
    int M, int Nout, int act, int totalWaves) {
  int wid = (blockIdx.x * 256 + threadIdx.x) >> 5;
  int lane = threadIdx.x & 31;
  if (wid >= totalWaves) return;
  int ngroups = Nout >> 6;
  int mt = wid / ngroups;
  int n0 = (wid - mt * ngroups) << 6;
  int m0 = mt << 4;

  // A k order used here (contiguous, same as B): lanes 0-15 -> K 0..15; lanes 16-31 -> K 16..31
  const bf16_t* Arow = A + (size_t)(m0 + (lane & 15)) * HID + ((lane >> 4) << 4);
  // B k order used here (contiguous, same as A): lanes 0-15 -> K 0..15; lanes 16-31 -> K 16..31 (contiguous in Wt)
  const bf16_t* Bcol = Wt + (size_t)(n0 + (lane & 15)) * HID + ((lane >> 4) << 4);

  v8f acc0 = {}, acc1 = {}, acc2 = {}, acc3 = {};
  for (int k0 = 0; k0 < HID; k0 += 32) {
    v8bf alo = *(const v8bf*)(Arow + k0);
    v8bf ahi = *(const v8bf*)(Arow + k0 + 8);
    v16bf a = __builtin_shufflevector(alo, ahi, 0, 1, 2, 3, 4, 5, 6, 7,
                                      8, 9, 10, 11, 12, 13, 14, 15);
    v16bf b0 = *(const v16bf*)(Bcol + k0);
    v16bf b1 = *(const v16bf*)(Bcol + 16 * HID + k0);
    v16bf b2 = *(const v16bf*)(Bcol + 32 * HID + k0);
    v16bf b3 = *(const v16bf*)(Bcol + 48 * HID + k0);
    __builtin_prefetch(Arow + k0 + 128, 0, 0);  // global_prefetch next A chunk
    acc0 = __builtin_amdgcn_wmma_f32_16x16x32_bf16(false, a, false, b0, (short)0, acc0, false, false);
    acc1 = __builtin_amdgcn_wmma_f32_16x16x32_bf16(false, a, false, b1, (short)0, acc1, false, false);
    acc2 = __builtin_amdgcn_wmma_f32_16x16x32_bf16(false, a, false, b2, (short)0, acc2, false, false);
    acc3 = __builtin_amdgcn_wmma_f32_16x16x32_bf16(false, a, false, b3, (short)0, acc3, false, false);
  }
  // C/D layout: VGPR i -> row m0 + 8*(lane>>4) + i, col n0 + (lane&15)
  int cn = lane & 15;
  int mr = m0 + ((lane >> 4) << 3);
  v8f accs[4] = {acc0, acc1, acc2, acc3};
#pragma unroll
  for (int j = 0; j < 4; ++j) {
    int col = n0 + j * 16 + cn;
    float bvv = bias ? bias[col] : 0.f;
#pragma unroll
    for (int i = 0; i < 8; ++i) {
      float v = accs[j][i] + bvv;
      if (act) v = eluf(v);
      C[(size_t)(mr + i) * Nout + col] = v;
    }
  }
}

// ---------- attention ----------
// wave per edge: alpha[e][h] = sum_c q[dst]*(k[src]+e) * 1/sqrt(C); seg-max via encoded atomicMax
__global__ void k_attn_score(const float* __restrict__ qf, const float* __restrict__ kf,
                             const float* __restrict__ ef, const int* __restrict__ src,
                             const int* __restrict__ dst, float* __restrict__ araw,
                             unsigned* __restrict__ nmax) {
  int e = (blockIdx.x * 256 + threadIdx.x) >> 5;
  int lane = threadIdx.x & 31;
  if (e >= NE) return;
  int s = src[e], d = dst[e];
  const float* qd = qf + (size_t)d * HID;
  const float* ks = kf + (size_t)s * HID;
  const float* ee = ef + (size_t)e * HID;
#pragma unroll
  for (int h = 0; h < NHEADS; ++h) {
    int b = h * 64 + lane;
    float p = qd[b] * (ks[b] + ee[b]) + qd[b + 32] * (ks[b + 32] + ee[b + 32]);
    p = wred(p);
    if (lane == 0) {
      float al = p * INV_SQRT_C;
      araw[(size_t)e * NHEADS + h] = al;
      atomicMax(&nmax[(size_t)d * NHEADS + h], encf(al));
    }
  }
}
__global__ void k_attn_exp(const float* __restrict__ araw, const int* __restrict__ dst,
                           const unsigned* __restrict__ nmax, float* __restrict__ aexp,
                           float* __restrict__ nsum) {
  int t = blockIdx.x * 256 + threadIdx.x;
  if (t >= NE * NHEADS) return;
  int e = t >> 3, h = t & 7;
  int d = dst[e];
  float ex = __expf(araw[t] - decf(nmax[(size_t)d * NHEADS + h]));
  aexp[t] = ex;
  atomicAdd(&nsum[(size_t)d * NHEADS + h], ex);
}
// wave per edge: agg[dst] += (v[src]+e) * alpha
__global__ void k_attn_agg(const float* __restrict__ vf, const float* __restrict__ ef,
                           const float* __restrict__ aexp, const float* __restrict__ nsum,
                           const int* __restrict__ src, const int* __restrict__ dst,
                           float* __restrict__ agg) {
  int e = (blockIdx.x * 256 + threadIdx.x) >> 5;
  int lane = threadIdx.x & 31;
  if (e >= NE) return;
  int s = src[e], d = dst[e];
  float wv = 0.f;
  if (lane < NHEADS)
    wv = aexp[(size_t)e * NHEADS + lane] / (nsum[(size_t)d * NHEADS + lane] + 1e-16f);
  const float* vs = vf + (size_t)s * HID;
  const float* ee = ef + (size_t)e * HID;
  float* ad = agg + (size_t)d * HID;
#pragma unroll
  for (int i = 0; i < 16; ++i) {
    int idx = i * 32 + lane;
    float wh = __shfl(wv, idx >> 6, 32);
    atomicAdd(&ad[idx], (vs[idx] + ee[idx]) * wh);
  }
}

// ---------- gated combine + ELU + global-LN stats (wave per node) ----------
__global__ void k_node_update(const float* __restrict__ agg, const float* __restrict__ xr,
                              const float* __restrict__ Wb, float* __restrict__ tmp,
                              float* __restrict__ stats) {
  int n = (blockIdx.x * 256 + threadIdx.x) >> 5;
  int lane = threadIdx.x & 31;
  if (n >= NN) return;
  const float* ag = agg + (size_t)n * HID;
  const float* xp = xr + (size_t)n * HID;
  float dot = 0.f;
  for (int j = lane; j < HID; j += 32) {
    float a = ag[j], x = xp[j];
    dot += a * Wb[j] + x * Wb[HID + j] + (a - x) * Wb[2 * HID + j];
  }
  dot = wred(dot);
  float beta = 1.f / (1.f + __expf(-dot));
  float s = 0.f, s2 = 0.f;
  for (int j = lane; j < HID; j += 32) {
    float o = beta * xp[j] + (1.f - beta) * ag[j];
    o = eluf(o);
    tmp[(size_t)n * HID + j] = o;
    s += o;
    s2 += o * o;
  }
  s = wred(s);
  s2 = wred(s2);
  if (lane == 0) {
    atomicAdd(&stats[0], s);
    atomicAdd(&stats[1], s2);
  }
}
__global__ void k_finalize_stats(float* stats, float cnt) {
  float mean = stats[0] / cnt;
  float var = stats[1] / cnt - mean * mean;
  stats[2] = mean;
  stats[3] = 1.f / (sqrtf(fmaxf(var, 0.f)) + 1e-5f);
}
__global__ void k_apply_ln(const float* __restrict__ tmp, const float* __restrict__ stats,
                           const float* __restrict__ gamma, const float* __restrict__ betaln,
                           float* __restrict__ h, bf16_t* __restrict__ hb,
                           float* __restrict__ fin, int layer) {
  int t = blockIdx.x * 256 + threadIdx.x;
  if (t >= NN * HID) return;
  int d = t & 511;
  float v = (tmp[t] - stats[2]) * stats[3] * gamma[d] + betaln[d];
  float hn = h[t] + v;
  h[t] = hn;
  hb[t] = f2bf(hn);
  fin[t] = (layer == 0) ? hn : fmaxf(fin[t], hn);
}

// ---------- global attention pooling ----------
__global__ void k_gate_dot(const float* __restrict__ g1, const float* __restrict__ Wg2,
                           const float* __restrict__ bg2, float* __restrict__ gate) {
  int n = (blockIdx.x * 256 + threadIdx.x) >> 5;
  int lane = threadIdx.x & 31;
  if (n >= NN) return;
  float dot = 0.f;
  for (int j = lane; j < 256; j += 32) dot += g1[(size_t)n * 256 + j] * Wg2[j];
  dot = wred(dot);
  if (lane == 0) gate[n] = dot + bg2[0];
}
__global__ void k_gate_max(const float* __restrict__ gate, const int* __restrict__ batch,
                           unsigned* __restrict__ gmax) {
  int n = blockIdx.x * 256 + threadIdx.x;
  if (n >= NN) return;
  atomicMax(&gmax[batch[n]], encf(gate[n]));
}
__global__ void k_gate_exp(const float* __restrict__ gate, const int* __restrict__ batch,
                           const unsigned* __restrict__ gmax, float* __restrict__ gexp,
                           float* __restrict__ gsum) {
  int n = blockIdx.x * 256 + threadIdx.x;
  if (n >= NN) return;
  int b = batch[n];
  float ex = __expf(gate[n] - decf(gmax[b]));
  gexp[n] = ex;
  atomicAdd(&gsum[b], ex);
}
__global__ void k_pool(const float* __restrict__ gexp, const float* __restrict__ gsum,
                       const int* __restrict__ batch, const float* __restrict__ fin,
                       float* __restrict__ pooled) {
  int t = blockIdx.x * 256 + threadIdx.x;
  if (t >= NN * HID) return;
  int n = t >> 9, d = t & 511;
  int b = batch[n];
  float w = gexp[n] / (gsum[b] + 1e-16f);
  atomicAdd(&pooled[(size_t)b * HID + d], w * fin[t]);
}

// ---------- scalar context ----------
__global__ void k_sctx1(const float* __restrict__ sc, const float* __restrict__ Ws1,
                        const float* __restrict__ bs1, float* __restrict__ s1) {
  int t = blockIdx.x * 256 + threadIdx.x;
  if (t >= NG * HID) return;
  int g = t >> 9, d = t & 511;
  float acc = bs1[d];
#pragma unroll
  for (int i = 0; i < 5; ++i) acc += sc[g * 5 + i] * Ws1[i * HID + d];
  s1[t] = eluf(acc);
}
__global__ void k_sctx2(const float* __restrict__ s1, const float* __restrict__ Ws2,
                        const float* __restrict__ bs2, float* __restrict__ pooled) {
  int t = blockIdx.x * 256 + threadIdx.x;
  if (t >= NG * HID) return;
  int g = t >> 9, d = t & 511;
  float acc = bs2[d];
  for (int k = 0; k < HID; ++k) acc += s1[(size_t)g * HID + k] * Ws2[(size_t)k * HID + d];
  pooled[t] += acc;
}

// ---------- readout heads (tiny dense stages, T=4) ----------
__global__ void k_head(const float* __restrict__ in, const float* __restrict__ W,
                       const float* __restrict__ b, float* __restrict__ out,
                       int G, int K, int J, int act, int shared_in) {
  int t = blockIdx.x * 256 + threadIdx.x;
  if (t >= NTGT * G * J) return;
  int tt = t / (G * J);
  int r = t - tt * (G * J);
  int g = r / J, j = r - g * J;
  const float* ip = shared_in ? in + (size_t)g * K : in + ((size_t)tt * G + g) * K;
  float acc = b[tt * J + j];
  for (int k = 0; k < K; ++k) acc += ip[k] * W[((size_t)tt * K + k) * J + j];
  if (act) acc = eluf(acc);
  out[((size_t)tt * G + g) * J + j] = acc;
}
__global__ void k_head_final(const float* __restrict__ t3, const float* __restrict__ Wh4,
                             const float* __restrict__ bh4, float* __restrict__ out) {
  int t = blockIdx.x * 64 + threadIdx.x;
  if (t >= NG * NTGT) return;
  int g = t / NTGT, tt = t - g * NTGT;
  float acc = bh4[tt];
  for (int k = 0; k < 64; ++k) acc += t3[((size_t)tt * NG + g) * 64 + k] * Wh4[tt * 64 + k];
  out[g * NTGT + tt] = acc;  // [G, NTGT]
}

// ---------- host ----------
extern "C" void kernel_launch(void* const* d_in, const int* in_sizes, int n_in,
                              void* d_out, int out_size, void* d_ws, size_t ws_size,
                              hipStream_t stream) {
  (void)in_sizes; (void)n_in; (void)out_size; (void)ws_size;
  const int* x_nodes    = (const int*)d_in[0];
  const int* edge_attr  = (const int*)d_in[1];
  const int* edge_index = (const int*)d_in[2];
  const int* batch_vec  = (const int*)d_in[3];
  const float* scalars  = (const float*)d_in[4];
  const float* node_emb = (const float*)d_in[5];
  const float* edge_emb = (const float*)d_in[6];
  const float* Wq = (const float*)d_in[7];
  const float* bq = (const float*)d_in[8];
  const float* Wk = (const float*)d_in[9];
  const float* bk = (const float*)d_in[10];
  const float* Wv = (const float*)d_in[11];
  const float* bv = (const float*)d_in[12];
  const float* We = (const float*)d_in[13];
  const float* Wskip = (const float*)d_in[14];
  const float* bskip = (const float*)d_in[15];
  const float* Wbeta = (const float*)d_in[16];
  const float* gammaP = (const float*)d_in[17];
  const float* beta_ln = (const float*)d_in[18];
  const float* Wg1 = (const float*)d_in[19];
  const float* bg1 = (const float*)d_in[20];
  const float* Wg2 = (const float*)d_in[21];
  const float* bg2 = (const float*)d_in[22];
  const float* Ws1 = (const float*)d_in[23];
  const float* bs1 = (const float*)d_in[24];
  const float* Ws2 = (const float*)d_in[25];
  const float* bs2 = (const float*)d_in[26];
  const float* Wh1 = (const float*)d_in[27];
  const float* bh1 = (const float*)d_in[28];
  const float* Wh2 = (const float*)d_in[29];
  const float* bh2 = (const float*)d_in[30];
  const float* Wh3 = (const float*)d_in[31];
  const float* bh3 = (const float*)d_in[32];
  const float* Wh4 = (const float*)d_in[33];
  const float* bh4 = (const float*)d_in[34];
  const int* srcI = edge_index;
  const int* dstI = edge_index + NE;

  char* wsp = (char*)d_ws;
  auto alloc = [&](size_t bytes) -> char* {
    char* p = wsp;
    wsp += (bytes + 255) & ~(size_t)255;
    return p;
  };
  const size_t WSZ = (size_t)HID * HID;  // elements per weight matrix
  bf16_t* wTq = (bf16_t*)alloc(NLAYERS * WSZ * 2);
  bf16_t* wTk = (bf16_t*)alloc(NLAYERS * WSZ * 2);
  bf16_t* wTv = (bf16_t*)alloc(NLAYERS * WSZ * 2);
  bf16_t* wTe = (bf16_t*)alloc(NLAYERS * WSZ * 2);
  bf16_t* wTs = (bf16_t*)alloc(NLAYERS * WSZ * 2);
  bf16_t* wTg1 = (bf16_t*)alloc((size_t)256 * HID * 2);
  float* h    = (float*)alloc((size_t)NN * HID * 4);
  bf16_t* hb  = (bf16_t*)alloc((size_t)NN * HID * 2);
  bf16_t* eb  = (bf16_t*)alloc((size_t)NE * HID * 2);
  float* qb   = (float*)alloc((size_t)NN * HID * 4);
  float* kb_  = (float*)alloc((size_t)NN * HID * 4);
  float* vb_  = (float*)alloc((size_t)NN * HID * 4);
  float* xr   = (float*)alloc((size_t)NN * HID * 4);
  float* eF   = (float*)alloc((size_t)NE * HID * 4);
  float* agg  = (float*)alloc((size_t)NN * HID * 4);
  float* tmp  = (float*)alloc((size_t)NN * HID * 4);
  float* fin  = (float*)alloc((size_t)NN * HID * 4);
  bf16_t* finb = (bf16_t*)alloc((size_t)NN * HID * 2);
  float* araw = (float*)alloc((size_t)NE * NHEADS * 4);
  float* aexp = (float*)alloc((size_t)NE * NHEADS * 4);
  unsigned* nmax = (unsigned*)alloc((size_t)NN * NHEADS * 4);
  float* nsum = (float*)alloc((size_t)NN * NHEADS * 4);
  float* stats = (float*)alloc(64);
  float* g1   = (float*)alloc((size_t)NN * 256 * 4);
  float* gate = (float*)alloc((size_t)NN * 4);
  float* gexp = (float*)alloc((size_t)NN * 4);
  unsigned* gmax = (unsigned*)alloc(NG * 4);
  float* gsum = (float*)alloc(NG * 4);
  float* pooled = (float*)alloc((size_t)NG * HID * 4);
  float* s1   = (float*)alloc((size_t)NG * HID * 4);
  float* t1   = (float*)alloc((size_t)NTGT * NG * 256 * 4);
  float* t2   = (float*)alloc((size_t)NTGT * NG * 128 * 4);
  float* t3   = (float*)alloc((size_t)NTGT * NG * 64 * 4);

  auto cdiv = [](long a, long b) { return (int)((a + b - 1) / b); };
  const unsigned ENC_NEG_INF = 0x007FFFFFu;  // encf(-inf)

  // ---- weight convert+transpose (fp32 -> bf16^T) ----
  for (int l = 0; l < NLAYERS; ++l) {
    int blk = cdiv(HID * HID, 256);
    k_cvt_wT<<<blk, 256, 0, stream>>>(Wq + l * WSZ, wTq + l * WSZ, HID, HID);
    k_cvt_wT<<<blk, 256, 0, stream>>>(Wk + l * WSZ, wTk + l * WSZ, HID, HID);
    k_cvt_wT<<<blk, 256, 0, stream>>>(Wv + l * WSZ, wTv + l * WSZ, HID, HID);
    k_cvt_wT<<<blk, 256, 0, stream>>>(We + l * WSZ, wTe + l * WSZ, HID, HID);
    k_cvt_wT<<<blk, 256, 0, stream>>>(Wskip + l * WSZ, wTs + l * WSZ, HID, HID);
  }
  k_cvt_wT<<<cdiv(HID * 256, 256), 256, 0, stream>>>(Wg1, wTg1, HID, 256);

  // ---- embeddings ----
  k_embed_nodes<<<cdiv((long)NN * HID, 256), 256, 0, stream>>>(x_nodes, node_emb, h, hb);
  k_embed_edges<<<cdiv((long)NE * HID, 256), 256, 0, stream>>>(edge_attr, edge_emb, eb);

  // ---- GNN layers ----
  const int twN = (NN / 16) * (HID / 64);   // node GEMM waves
  const int twE = (NE / 16) * (HID / 64);   // edge GEMM waves
  for (int l = 0; l < NLAYERS; ++l) {
    size_t wo = (size_t)l * WSZ;
    k_gemm_bf16<<<cdiv((long)twN * 32, 256), 256, 0, stream>>>(hb, wTq + wo, bq + l * HID, qb, NN, HID, 0, twN);
    k_gemm_bf16<<<cdiv((long)twN * 32, 256), 256, 0, stream>>>(hb, wTk + wo, bk + l * HID, kb_, NN, HID, 0, twN);
    k_gemm_bf16<<<cdiv((long)twN * 32, 256), 256, 0, stream>>>(hb, wTv + wo, bv + l * HID, vb_, NN, HID, 0, twN);
    k_gemm_bf16<<<cdiv((long)twN * 32, 256), 256, 0, stream>>>(hb, wTs + wo, bskip + l * HID, xr, NN, HID, 0, twN);
    k_gemm_bf16<<<cdiv((long)twE * 32, 256), 256, 0, stream>>>(eb, wTe + wo, nullptr, eF, NE, HID, 0, twE);

    k_fill_f32<<<cdiv((long)NN * HID, 256), 256, 0, stream>>>(agg, 0.f, NN * HID);
    k_fill_u32<<<cdiv(NN * NHEADS, 256), 256, 0, stream>>>(nmax, ENC_NEG_INF, NN * NHEADS);
    k_fill_f32<<<cdiv(NN * NHEADS, 256), 256, 0, stream>>>(nsum, 0.f, NN * NHEADS);
    k_fill_f32<<<1, 256, 0, stream>>>(stats, 0.f, 8);

    k_attn_score<<<cdiv((long)NE * 32, 256), 256, 0, stream>>>(qb, kb_, eF, srcI, dstI, araw, nmax);
    k_attn_exp<<<cdiv((long)NE * NHEADS, 256), 256, 0, stream>>>(araw, dstI, nmax, aexp, nsum);
    k_attn_agg<<<cdiv((long)NE * 32, 256), 256, 0, stream>>>(vb_, eF, aexp, nsum, srcI, dstI, agg);

    k_node_update<<<cdiv((long)NN * 32, 256), 256, 0, stream>>>(agg, xr, Wbeta + (size_t)l * 3 * HID, tmp, stats);
    k_finalize_stats<<<1, 1, 0, stream>>>(stats, (float)((size_t)NN * HID));
    k_apply_ln<<<cdiv((long)NN * HID, 256), 256, 0, stream>>>(tmp, stats, gammaP + l * HID,
                                                              beta_ln + l * HID, h, hb, fin, l);
  }

  // ---- global attention pooling ----
  k_cvt_bf16<<<cdiv((long)NN * HID, 256), 256, 0, stream>>>(fin, finb, NN * HID);
  const int twG = (NN / 16) * (256 / 64);
  k_gemm_bf16<<<cdiv((long)twG * 32, 256), 256, 0, stream>>>(finb, wTg1, bg1, g1, NN, 256, 1, twG);
  k_gate_dot<<<cdiv((long)NN * 32, 256), 256, 0, stream>>>(g1, Wg2, bg2, gate);
  k_fill_u32<<<1, 256, 0, stream>>>(gmax, ENC_NEG_INF, NG);
  k_fill_f32<<<1, 256, 0, stream>>>(gsum, 0.f, NG);
  k_fill_f32<<<cdiv(NG * HID, 256), 256, 0, stream>>>(pooled, 0.f, NG * HID);
  k_gate_max<<<cdiv(NN, 256), 256, 0, stream>>>(gate, batch_vec, gmax);
  k_gate_exp<<<cdiv(NN, 256), 256, 0, stream>>>(gate, batch_vec, gmax, gexp, gsum);
  k_pool<<<cdiv((long)NN * HID, 256), 256, 0, stream>>>(gexp, gsum, batch_vec, fin, pooled);

  // ---- scalar context ----
  k_sctx1<<<cdiv(NG * HID, 256), 256, 0, stream>>>(scalars, Ws1, bs1, s1);
  k_sctx2<<<cdiv(NG * HID, 256), 256, 0, stream>>>(s1, Ws2, bs2, pooled);

  // ---- readout heads ----
  k_head<<<cdiv(NTGT * NG * 256, 256), 256, 0, stream>>>(pooled, Wh1, bh1, t1, NG, HID, 256, 1, 1);
  k_head<<<cdiv(NTGT * NG * 128, 256), 256, 0, stream>>>(t1, Wh2, bh2, t2, NG, 256, 128, 1, 0);
  k_head<<<cdiv(NTGT * NG * 64, 256), 256, 0, stream>>>(t2, Wh3, bh3, t3, NG, 128, 64, 1, 0);
  k_head_final<<<1, 64, 0, stream>>>(t3, Wh4, bh4, (float*)d_out);
}
